// LSTM3lr_EncoderCell_76690936037398
// MI455X (gfx1250) — matrix operand order repaired and checked by reading, not compiled, not run on hardware
//
#include <hip/hip_runtime.h>

// ---------------------------------------------------------------------------
// 3-layer LSTM (B=256, H=1000, T=49, D=192) for gfx1250 / MI455X.
//  - f16 weights (gate-interleaved rows n' = 4*unit+gate, N padded to 4096,
//    K padded to 1024) resident in the 192MB L2; fp32 WMMA accumulation.
//  - One fused kernel per (t,layer): WMMA GEMM + bias + LSTM cell update.
//  - Software-pipelined fragment loads: next k-step's 10 b128 loads are issued
//    before the current k-step's 4 WMMAs, so waits become partial
//    (s_wait_loadcnt N) instead of full barriers before every WMMA.
// ---------------------------------------------------------------------------

typedef __attribute__((ext_vector_type(16))) _Float16 v16h;
typedef __attribute__((ext_vector_type(8)))  float    v8f;

#define HIDN   1000
#define BATCH  256
#define TSTEPS 49
#define DIN    192
#define NPAD   4096   // 4*1024 interleaved gate rows
#define KPAD   1024

union FragU { uint4 q[2]; v16h v; };

// p is the lane-resolved pointer at the k0 position.
// A fragment (16x32 MxK): per-lane chunks at +0 and +8 halves.
__device__ __forceinline__ v16h ld_a(const _Float16* __restrict__ p) {
  FragU u;
  u.q[0] = *(const uint4*)(p);
  u.q[1] = *(const uint4*)(p + 8);
  return u.v;
}
// B fragment (32x16 KxN) from row-major W[N][K]: per-lane chunks at +0, +8.
__device__ __forceinline__ v16h ld_b(const _Float16* __restrict__ p) {
  FragU u;
  u.q[0] = *(const uint4*)(p);
  u.q[1] = *(const uint4*)(p + 8);
  return u.v;
}

__device__ __forceinline__ float sigm_f(float x) { return 1.f / (1.f + __expf(-x)); }
__device__ __forceinline__ float tanh_f(float x) { return 2.f / (1.f + __expf(-2.f * x)) - 1.f; }

#define WMMA(A, Bf, C) __builtin_amdgcn_wmma_f32_16x16x32_f16( \
    false, (A), false, (Bf), (short)0, (C), false, false)

// ---------------------------------------------------------------------------
// Fused LSTM step. Grid: (16 M-blocks, 16 N-blocks), block = 128 (4 waves).
// Wave w: M16 x N64 tile group at nb = (by*4+w)*64.
// ---------------------------------------------------------------------------
__global__ __launch_bounds__(128)
void lstm_step_kernel(const _Float16* __restrict__ X, int Kx,
                      const _Float16* __restrict__ Wih,
                      const _Float16* __restrict__ Hin,
                      const _Float16* __restrict__ Whh,
                      const float*    __restrict__ bias,
                      float*          __restrict__ Cst,
                      _Float16*       __restrict__ Hout,
                      float*          __restrict__ out,
                      int t, int layer) {
  __shared__ float smem[4][256];
  const int lane = threadIdx.x & 31;
  const int wave = threadIdx.x >> 5;
  const int m0 = blockIdx.x * 16;
  const int nb = (blockIdx.y * 4 + wave) * 64;
  const int r = lane & 15, g = lane >> 4;

  v8f acc[4] = {v8f{}, v8f{}, v8f{}, v8f{}};

  // Software-pipelined K segment: preload k0, then load k0+32 before the
  // 4 WMMAs consuming k0's fragments.
#define SEGMENT(Abase, Wbase, K)                                              \
  {                                                                           \
    const _Float16* pa  = (Abase) + (size_t)(m0 + r) * (K) + 16 * g;          \
    const _Float16* pb0 = (Wbase) + (size_t)(nb +  0 + r) * (K) + 16 * g;     \
    const _Float16* pb1 = (Wbase) + (size_t)(nb + 16 + r) * (K) + 16 * g;     \
    const _Float16* pb2 = (Wbase) + (size_t)(nb + 32 + r) * (K) + 16 * g;     \
    const _Float16* pb3 = (Wbase) + (size_t)(nb + 48 + r) * (K) + 16 * g;     \
    v16h a  = ld_a(pa);                                                       \
    v16h b0 = ld_b(pb0), b1 = ld_b(pb1), b2 = ld_b(pb2), b3 = ld_b(pb3);      \
    for (int k0 = 32; k0 < (K); k0 += 32) {                                   \
      pa += 32; pb0 += 32; pb1 += 32; pb2 += 32; pb3 += 32;                   \
      v16h an  = ld_a(pa);                                                    \
      v16h c0 = ld_b(pb0), c1 = ld_b(pb1), c2 = ld_b(pb2), c3 = ld_b(pb3);    \
      acc[0] = WMMA(a, b0, acc[0]);                                           \
      acc[1] = WMMA(a, b1, acc[1]);                                           \
      acc[2] = WMMA(a, b2, acc[2]);                                           \
      acc[3] = WMMA(a, b3, acc[3]);                                           \
      a = an; b0 = c0; b1 = c1; b2 = c2; b3 = c3;                             \
    }                                                                         \
    acc[0] = WMMA(a, b0, acc[0]);                                             \
    acc[1] = WMMA(a, b1, acc[1]);                                             \
    acc[2] = WMMA(a, b2, acc[2]);                                             \
    acc[3] = WMMA(a, b3, acc[3]);                                             \
  }

  SEGMENT(X, Wih, Kx)       // input contribution
  SEGMENT(Hin, Whh, KPAD)   // recurrent contribution
#undef SEGMENT

  // --- Epilogue: bias, stage tile to LDS, LSTM cell update ---
#pragma unroll
  for (int j = 0; j < 4; ++j) {
    const int n16 = nb + 16 * j;               // n' base of this 16-wide tile
    const float bv = bias[n16 + r];            // bias fixed per n' column
#pragma unroll
    for (int d = 0; d < 8; ++d)
      smem[wave][(d + 8 * g) * 16 + r] = acc[j][d] + bv;  // [m_local][n_local]
    __syncthreads();

    // Tile holds gates i,f,g,o (n_local%4) of units u0..u0+3, 16 batch rows.
#pragma unroll
    for (int it = 0; it < 2; ++it) {
      const int p  = it * 32 + lane;           // 0..63 cells
      const int ml = p >> 2;                   // batch row within tile
      const int ul = p & 3;                    // unit within tile
      const int u  = (n16 >> 2) + ul;          // global hidden unit
      if (u < HIDN) {
        const float iv = smem[wave][ml * 16 + 4 * ul + 0];
        const float fv = smem[wave][ml * 16 + 4 * ul + 1];
        const float gv = smem[wave][ml * 16 + 4 * ul + 2];
        const float ov = smem[wave][ml * 16 + 4 * ul + 3];
        const int brow = m0 + ml;
        const size_t ci = (size_t)brow * KPAD + u;
        const float c_old = Cst[ci];
        const float c_new = sigm_f(fv) * c_old + sigm_f(iv) * tanh_f(gv);
        const float h     = sigm_f(ov) * tanh_f(c_new);
        Cst[ci]  = c_new;
        Hout[ci] = (_Float16)h;
        out[((size_t)(layer * BATCH + brow) * TSTEPS + t) * HIDN + u] = h;
      }
    }
    __syncthreads();
  }
}

// ---------------------------------------------------------------------------
// One-time conversion kernels
// ---------------------------------------------------------------------------

// src: fp32 [4*1000][Kin] (PyTorch gate-stacked rows). dst: f16 [4096][Kpad],
// row n' = 4*unit + gate, zero padding for unit>=1000 or k>=Kin.
__global__ void convert_weight(const float* __restrict__ src,
                               _Float16* __restrict__ dst, int Kin, int Kpad) {
  size_t idx = (size_t)blockIdx.x * blockDim.x + threadIdx.x;
  size_t total = (size_t)NPAD * Kpad;
  if (idx >= total) return;
  int k  = (int)(idx % Kpad);
  int np = (int)(idx / Kpad);
  int u = np >> 2, gi = np & 3;
  float v = 0.f;
  if (u < HIDN && k < Kin) v = src[(size_t)(gi * HIDN + u) * Kin + k];
  dst[idx] = (_Float16)v;
}

__global__ void make_bias(const float* __restrict__ bih,
                          const float* __restrict__ bhh,
                          float* __restrict__ dst) {
  int np = blockIdx.x * blockDim.x + threadIdx.x;
  if (np >= NPAD) return;
  int u = np >> 2, gi = np & 3;
  dst[np] = (u < HIDN) ? (bih[gi * HIDN + u] + bhh[gi * HIDN + u]) : 0.f;
}

// enc_in [B][T][D] fp32  ->  X0 [T][B][D] f16 (time-major per-step slices)
__global__ void convert_x(const float* __restrict__ enc,
                          _Float16* __restrict__ X0) {
  size_t idx = (size_t)blockIdx.x * blockDim.x + threadIdx.x;
  size_t total = (size_t)TSTEPS * BATCH * DIN;
  if (idx >= total) return;
  int d   = (int)(idx % DIN);
  int rem = (int)(idx / DIN);
  int b   = rem % BATCH;
  int t   = rem / BATCH;
  X0[idx] = (_Float16)enc[((size_t)b * TSTEPS + t) * DIN + d];
}

// ---------------------------------------------------------------------------
extern "C" void kernel_launch(void* const* d_in, const int* in_sizes, int n_in,
                              void* d_out, int out_size, void* d_ws, size_t ws_size,
                              hipStream_t stream) {
  (void)in_sizes; (void)n_in; (void)out_size; (void)ws_size;

  const float* enc = (const float*)d_in[0];
  const float* Wih_src[3] = {(const float*)d_in[1], (const float*)d_in[5], (const float*)d_in[9]};
  const float* Whh_src[3] = {(const float*)d_in[2], (const float*)d_in[6], (const float*)d_in[10]};
  const float* bih_src[3] = {(const float*)d_in[3], (const float*)d_in[7], (const float*)d_in[11]};
  const float* bhh_src[3] = {(const float*)d_in[4], (const float*)d_in[8], (const float*)d_in[12]};

  // ---- carve workspace (all sizes are multiples of 256 bytes) ----
  char* ws = (char*)d_ws;
  size_t off = 0;
  const int KpadIh[3] = {DIN, KPAD, KPAD};
  const int KinIh[3]  = {DIN, HIDN, HIDN};
  _Float16* Wih[3]; _Float16* Whh[3];
  for (int l = 0; l < 3; ++l) {
    Wih[l] = (_Float16*)(ws + off); off += (size_t)NPAD * KpadIh[l] * sizeof(_Float16);
    Whh[l] = (_Float16*)(ws + off); off += (size_t)NPAD * KPAD      * sizeof(_Float16);
  }
  float* biasws = (float*)(ws + off);     off += (size_t)3 * NPAD * sizeof(float);
  _Float16* X0  = (_Float16*)(ws + off);  off += (size_t)TSTEPS * BATCH * DIN * sizeof(_Float16);
  _Float16* Hbuf = (_Float16*)(ws + off); // [3][2][BATCH*KPAD]
  size_t hElems = (size_t)3 * 2 * BATCH * KPAD;
  off += hElems * sizeof(_Float16);
  float* Cbuf = (float*)(ws + off);       // [3][BATCH*KPAD]
  size_t cElems = (size_t)3 * BATCH * KPAD;
  off += cElems * sizeof(float);

  // ---- one-time per-launch setup ----
  for (int l = 0; l < 3; ++l) {
    size_t nIh = (size_t)NPAD * KpadIh[l];
    convert_weight<<<(unsigned)((nIh + 255) / 256), 256, 0, stream>>>(
        Wih_src[l], Wih[l], KinIh[l], KpadIh[l]);
    size_t nHh = (size_t)NPAD * KPAD;
    convert_weight<<<(unsigned)((nHh + 255) / 256), 256, 0, stream>>>(
        Whh_src[l], Whh[l], HIDN, KPAD);
    make_bias<<<NPAD / 256, 256, 0, stream>>>(bih_src[l], bhh_src[l], biasws + (size_t)l * NPAD);
  }
  {
    size_t nX = (size_t)TSTEPS * BATCH * DIN;
    convert_x<<<(unsigned)((nX + 255) / 256), 256, 0, stream>>>(enc, X0);
  }
  hipMemsetAsync(Hbuf, 0, hElems * sizeof(_Float16), stream);
  hipMemsetAsync(Cbuf, 0, cElems * sizeof(float), stream);

  // ---- the time/layer recurrence: one fused kernel per (t, layer) ----
  const size_t hStride = (size_t)BATCH * KPAD;
  for (int t = 0; t < TSTEPS; ++t) {
    for (int l = 0; l < 3; ++l) {
      const _Float16* X;
      int Kx;
      if (l == 0) { X = X0 + (size_t)t * BATCH * DIN; Kx = DIN; }
      else        { X = Hbuf + ((size_t)(l - 1) * 2 + ((t + 1) & 1)) * hStride; Kx = KPAD; }
      const _Float16* Hin  = Hbuf + ((size_t)l * 2 + (t & 1)) * hStride;
      _Float16*       Hout = Hbuf + ((size_t)l * 2 + ((t + 1) & 1)) * hStride;
      lstm_step_kernel<<<dim3(16, 16), 128, 0, stream>>>(
          X, Kx, Wih[l], Hin, Whh[l], biasws + (size_t)l * NPAD,
          Cbuf + (size_t)l * hStride, Hout, (float*)d_out, t, l);
    }
  }
}
